// FeedforwardGNN_90503550861445
// MI455X (gfx1250) — matrix operand order repaired and checked by reading, not compiled, not run on hardware
//
#include <hip/hip_runtime.h>
#include <hip/hip_bf16.h>

typedef _Float16 v16h __attribute__((ext_vector_type(16)));
typedef float    v8f  __attribute__((ext_vector_type(8)));
typedef int      v4i  __attribute__((__vector_size__(16)));

#define DEV __device__ __forceinline__
#define AS1 __attribute__((address_space(1)))
#define AS3 __attribute__((address_space(3)))

constexpr int NN = 512;   // nodes
constexpr int DH = 64;    // hidden = message dim
constexpr int TT = 128;   // time steps
constexpr int EE = 8192;  // edges

// ---------------------------------------------------------------- WMMA helpers
DEV v8f wmma16(v16h a, v16h b, v8f c) {
  return __builtin_amdgcn_wmma_f32_16x16x32_f16(false, a, false, b, (short)0, c,
                                                false, false);
}

// A-matrix 16x32 f16 (ISA 7.12.2): lane row = lane&15, half = lane>>4.
// K pairs are contiguous -> fetch as 8 dwords (global_load_b32 / ds_load_b32).
DEV v16h load_a_f16(const _Float16* __restrict__ row, int half) {
  union { v16h v; uint32_t u[8]; } a;
#pragma unroll
  for (int p = 0; p < 8; ++p) {
    int k = ((p < 4) ? (2 * p) : (16 + 2 * (p - 4))) + half * 8;
    a.u[p] = *reinterpret_cast<const uint32_t*>(row + k);
  }
  return a.v;
}

DEV v16h load_a_f32(const float* __restrict__ row, int half) {
  v16h a;
#pragma unroll
  for (int p = 0; p < 8; ++p) {
    int k = ((p < 4) ? (2 * p) : (16 + 2 * (p - 4))) + half * 8;
    float2 f = *reinterpret_cast<const float2*>(row + k);
    a[2 * p]     = (_Float16)f.x;
    a[2 * p + 1] = (_Float16)f.y;
  }
  return a;
}

// B tiles live pre-swizzled: tile (kc,nc) stored as 32 lanes x 16 contiguous
// halfs. One wave reads its v16h as two ds_load_b128.
DEV v16h load_b_swz(const _Float16* __restrict__ base) {
  return *reinterpret_cast<const v16h*>(base);
}

DEV float silu(float x) { return x / (1.0f + __expf(-x)); }

// ----------------------------------------------------- async weight staging
#if __has_builtin(__builtin_amdgcn_global_load_async_to_lds_b128)
#define HAVE_ASYNC 1
#else
#define HAVE_ASYNC 0
#endif

DEV void stage_weights(const _Float16* __restrict__ w1,
                       const _Float16* __restrict__ w2,
                       _Float16* w1s, _Float16* w2s, int tid) {
#if HAVE_ASYNC
#pragma unroll
  for (int it = 0; it < 4; ++it) {      // W1: 8192 halfs = 1024 x 16B
    int off = (tid + it * 256) * 8;
    __builtin_amdgcn_global_load_async_to_lds_b128(
        (AS1 v4i*)(v4i*)(void*)(w1 + off),
        (AS3 v4i*)(v4i*)(void*)(w1s + off), 0, 0);
  }
#pragma unroll
  for (int it = 0; it < 2; ++it) {      // W2: 4096 halfs = 512 x 16B
    int off = (tid + it * 256) * 8;
    __builtin_amdgcn_global_load_async_to_lds_b128(
        (AS1 v4i*)(v4i*)(void*)(w2 + off),
        (AS3 v4i*)(v4i*)(void*)(w2s + off), 0, 0);
  }
#if __has_builtin(__builtin_amdgcn_s_wait_asynccnt)
  __builtin_amdgcn_s_wait_asynccnt(0);
#else
  asm volatile("s_wait_asynccnt 0" ::: "memory");
#endif
#else
#pragma unroll
  for (int it = 0; it < 4; ++it) {
    int off = tid + it * 256;
    reinterpret_cast<v4i*>(w1s)[off] = reinterpret_cast<const v4i*>(w1)[off];
  }
#pragma unroll
  for (int it = 0; it < 2; ++it) {
    int off = tid + it * 256;
    reinterpret_cast<v4i*>(w2s)[off] = reinterpret_cast<const v4i*>(w2)[off];
  }
#endif
}

// ---------------------------------------------------------------- small kernels
// f32 [K,64] weight -> f16 pre-swizzled WMMA-B layout:
// dst[((kc*4+nc)*32 + lane)*16 + j] = src[(kc*32 + (lane>>4)*8 + (j&7) + (j>>3)*16)*64 + nc*16 + (lane&15)]
__global__ void cvt_swz_kernel(const float* __restrict__ src,
                               _Float16* __restrict__ dst, int per, int total) {
  int i = blockIdx.x * blockDim.x + threadIdx.x;
  if (i >= total) return;
  int l  = i / per;          // layer
  int ii = i % per;
  int j    = ii & 15;
  int L    = (ii >> 4) & 31;
  int tile = ii >> 9;
  int nc = tile & 3, kc = tile >> 2;
  int k = kc * 32 + (L >> 4) * 8 + (j & 7) + (j >> 3) * 16;
  int n = nc * 16 + (L & 15);
  dst[i] = (_Float16)src[l * per + k * 64 + n];
}

__global__ void zero_kernel(float* __restrict__ p) {
  p[blockIdx.x * blockDim.x + threadIdx.x] = 0.0f;
}

// h[t,n,k] = t[t]*enc_w[0,k] + enc_w[1+n,k] + enc_b[k]  (one-hot matmul collapsed)
__global__ void enc_kernel(const float* __restrict__ tvec,
                           const float* __restrict__ enc_w,
                           const float* __restrict__ enc_b,
                           float* __restrict__ h32, _Float16* __restrict__ h16) {
  int i = blockIdx.x * blockDim.x + threadIdx.x;  // over T*N*DH
  int k = i & (DH - 1);
  int n = (i >> 6) & (NN - 1);
  int t = i >> 15;  // / (N*DH) = 32768
  float v = tvec[t] * enc_w[k] + enc_w[(1 + n) * DH + k] + enc_b[k];
  h32[i] = v;
  h16[i] = (_Float16)v;
}

__global__ void ro_kernel(const float* __restrict__ h32,
                          const float* __restrict__ row,
                          const float* __restrict__ rob, float* __restrict__ out) {
  int i = blockIdx.x * blockDim.x + threadIdx.x;  // over T*N
  const float* r = h32 + (size_t)i * DH;
  float s = rob[0];
#pragma unroll
  for (int k = 0; k < DH; ++k) s += r[k] * row[k];
  out[i] = s;
}

// ---------------------------------------------------------------- message kernel
// One wave = 16 edges. m = silu([h_src|h_dst] @ W1 + b1) @ W2 + b2, scatter to agg.
__global__ void __launch_bounds__(256) msg_kernel(
    const _Float16* __restrict__ h16, const int* __restrict__ esrc,
    const int* __restrict__ edst, const _Float16* __restrict__ w1,
    const float* __restrict__ b1, const _Float16* __restrict__ w2,
    const float* __restrict__ b2, float* __restrict__ agg) {
  __shared__ alignas(32) _Float16 w1s[16 * 32 * 16];   // 16 KiB
  __shared__ alignas(32) _Float16 w2s[8 * 32 * 16];    //  8 KiB
  __shared__ alignas(32) _Float16 hstage[8][16 * DH];  // 16 KiB
  const int wave = threadIdx.x >> 5;
  const int lane = threadIdx.x & 31;
  const int half = lane >> 4;
  const int col  = lane & 15;
  const int tile = blockIdx.x * 8 + wave;
  const int t  = tile >> 9;           // / (E/16 = 512)
  const int e0 = (tile & 511) * 16;

  stage_weights(w1, w2, w1s, w2s, threadIdx.x);
  __syncthreads();

  const int m  = col;                 // this lane's A row (edge)
  const int sn = esrc[e0 + m];
  const int dn = edst[e0 + m];
  const _Float16* rs = h16 + (size_t)(t * NN + sn) * DH;
  const _Float16* rd = h16 + (size_t)(t * NN + dn) * DH;

  // GEMM1: [16x128] x [128x64], K-chunks 0,1 = src rows, 2,3 = dst rows
  v8f acc[4];
#pragma unroll
  for (int nc = 0; nc < 4; ++nc) {
    float bv = b1[nc * 16 + col];
#pragma unroll
    for (int r = 0; r < 8; ++r) acc[nc][r] = bv;
  }
#pragma unroll
  for (int kc = 0; kc < 4; ++kc) {
    const _Float16* ar = (kc < 2) ? (rs + kc * 32) : (rd + (kc - 2) * 32);
    v16h a = load_a_f16(ar, half);
#pragma unroll
    for (int nc = 0; nc < 4; ++nc) {
      v16h b = load_b_swz(&w1s[((kc * 4 + nc) * 32 + lane) * 16]);
      acc[nc] = wmma16(a, b, acc[nc]);
    }
  }
  // SiLU + C->A re-layout through LDS (row-major 16x64 tile per wave)
#pragma unroll
  for (int nc = 0; nc < 4; ++nc)
#pragma unroll
    for (int r = 0; r < 8; ++r)
      hstage[wave][(r + 8 * half) * DH + nc * 16 + col] =
          (_Float16)silu(acc[nc][r]);
  __syncthreads();

  // GEMM2: [16x64] x [64x64]
  v8f acc2[4];
#pragma unroll
  for (int nc = 0; nc < 4; ++nc) {
    float bv = b2[nc * 16 + col];
#pragma unroll
    for (int r = 0; r < 8; ++r) acc2[nc][r] = bv;
  }
#pragma unroll
  for (int kc = 0; kc < 2; ++kc) {
    v16h a = load_a_f16(&hstage[wave][m * DH + kc * 32], half);
#pragma unroll
    for (int nc = 0; nc < 4; ++nc) {
      v16h b = load_b_swz(&w2s[((kc * 4 + nc) * 32 + lane) * 16]);
      acc2[nc] = wmma16(a, b, acc2[nc]);
    }
  }
  // scatter-add messages into agg[t, dst, :]
  int dnode[8];
#pragma unroll
  for (int r = 0; r < 8; ++r) dnode[r] = edst[e0 + r + 8 * half];
#pragma unroll
  for (int nc = 0; nc < 4; ++nc)
#pragma unroll
    for (int r = 0; r < 8; ++r)
      atomicAdd(&agg[(size_t)(t * NN + dnode[r]) * DH + nc * 16 + col],
                acc2[nc][r]);
}

// ---------------------------------------------------------------- update kernel
// One wave = 16 nodes. h += silu([h|agg] @ U1 + b1) @ U2 + b2
__global__ void __launch_bounds__(256) upd_kernel(
    _Float16* __restrict__ h16, const float* __restrict__ agg,
    const _Float16* __restrict__ w1, const float* __restrict__ b1,
    const _Float16* __restrict__ w2, const float* __restrict__ b2,
    float* __restrict__ h32) {
  __shared__ alignas(32) _Float16 w1s[16 * 32 * 16];
  __shared__ alignas(32) _Float16 w2s[8 * 32 * 16];
  __shared__ alignas(32) _Float16 hstage[8][16 * DH];
  const int wave = threadIdx.x >> 5;
  const int lane = threadIdx.x & 31;
  const int half = lane >> 4;
  const int col  = lane & 15;
  const int tile = blockIdx.x * 8 + wave;
  const int t  = tile >> 5;           // N/16 = 32 tiles per t
  const int n0 = (tile & 31) * 16;

  stage_weights(w1, w2, w1s, w2s, threadIdx.x);
  __syncthreads();

  const int m = col;
  const _Float16* rh = h16 + (size_t)(t * NN + n0 + m) * DH;
  const float*    ra = agg + (size_t)(t * NN + n0 + m) * DH;

  v8f acc[4];
#pragma unroll
  for (int nc = 0; nc < 4; ++nc) {
    float bv = b1[nc * 16 + col];
#pragma unroll
    for (int r = 0; r < 8; ++r) acc[nc][r] = bv;
  }
#pragma unroll
  for (int kc = 0; kc < 4; ++kc) {
    v16h a = (kc < 2) ? load_a_f16(rh + kc * 32, half)
                      : load_a_f32(ra + (kc - 2) * 32, half);
#pragma unroll
    for (int nc = 0; nc < 4; ++nc) {
      v16h b = load_b_swz(&w1s[((kc * 4 + nc) * 32 + lane) * 16]);
      acc[nc] = wmma16(a, b, acc[nc]);
    }
  }
#pragma unroll
  for (int nc = 0; nc < 4; ++nc)
#pragma unroll
    for (int r = 0; r < 8; ++r)
      hstage[wave][(r + 8 * half) * DH + nc * 16 + col] =
          (_Float16)silu(acc[nc][r]);
  __syncthreads();

  v8f acc2[4];
#pragma unroll
  for (int nc = 0; nc < 4; ++nc) {
    float bv = b2[nc * 16 + col];
#pragma unroll
    for (int r = 0; r < 8; ++r) acc2[nc][r] = bv;
  }
#pragma unroll
  for (int kc = 0; kc < 2; ++kc) {
    v16h a = load_a_f16(&hstage[wave][m * DH + kc * 32], half);
#pragma unroll
    for (int nc = 0; nc < 4; ++nc) {
      v16h b = load_b_swz(&w2s[((kc * 4 + nc) * 32 + lane) * 16]);
      acc2[nc] = wmma16(a, b, acc2[nc]);
    }
  }
  // residual + write back both precisions (rows of this tile are wave-exclusive)
#pragma unroll
  for (int nc = 0; nc < 4; ++nc)
#pragma unroll
    for (int r = 0; r < 8; ++r) {
      size_t idx = (size_t)(t * NN + n0 + r + 8 * half) * DH + nc * 16 + col;
      float nv = h32[idx] + acc2[nc][r];
      h32[idx] = nv;
      h16[idx] = (_Float16)nv;
    }
}

// ---------------------------------------------------------------- launch
extern "C" void kernel_launch(void* const* d_in, const int* in_sizes, int n_in,
                              void* d_out, int out_size, void* d_ws, size_t ws_size,
                              hipStream_t stream) {
  const float* t_in   = (const float*)d_in[0];
  const int*   edge   = (const int*)d_in[1];   // [2, E]
  const float* enc_w  = (const float*)d_in[2];
  const float* enc_b  = (const float*)d_in[3];
  const float* msg1_w = (const float*)d_in[4];
  const float* msg1_b = (const float*)d_in[5];
  const float* msg2_w = (const float*)d_in[6];
  const float* msg2_b = (const float*)d_in[7];
  const float* upd1_w = (const float*)d_in[8];
  const float* upd1_b = (const float*)d_in[9];
  const float* upd2_w = (const float*)d_in[10];
  const float* upd2_b = (const float*)d_in[11];
  const float* ro_w   = (const float*)d_in[12];
  const float* ro_b   = (const float*)d_in[13];

  char* ws = (char*)d_ws;
  const size_t HBYTES = (size_t)TT * NN * DH * 4;  // 16 MiB
  float*     h32 = (float*)ws;
  float*     agg = (float*)(ws + HBYTES);
  _Float16*  h16 = (_Float16*)(ws + 2 * HBYTES);
  _Float16*  w16 = (_Float16*)(ws + 2 * HBYTES + HBYTES / 2);
  _Float16* m1w = w16;                        // 3 * 128 * 64 (swizzled)
  _Float16* m2w = m1w + 3 * 2 * DH * DH;      // 3 * 64 * 64
  _Float16* u1w = m2w + 3 * DH * DH;          // 3 * 128 * 64
  _Float16* u2w = u1w + 3 * 2 * DH * DH;      // 3 * 64 * 64

  // f32 -> f16 weight conversion with WMMA-B pre-swizzle
  cvt_swz_kernel<<<(3 * 2 * DH * DH + 255) / 256, 256, 0, stream>>>(
      msg1_w, m1w, 2 * DH * DH, 3 * 2 * DH * DH);
  cvt_swz_kernel<<<(3 * DH * DH + 255) / 256, 256, 0, stream>>>(
      msg2_w, m2w, DH * DH, 3 * DH * DH);
  cvt_swz_kernel<<<(3 * 2 * DH * DH + 255) / 256, 256, 0, stream>>>(
      upd1_w, u1w, 2 * DH * DH, 3 * 2 * DH * DH);
  cvt_swz_kernel<<<(3 * DH * DH + 255) / 256, 256, 0, stream>>>(
      upd2_w, u2w, DH * DH, 3 * DH * DH);

  // encoder (one-hot matmul collapsed to broadcast-add)
  enc_kernel<<<(TT * NN * DH) / 256, 256, 0, stream>>>(t_in, enc_w, enc_b, h32,
                                                       h16);

  const int* esrc = edge;
  const int* edst = edge + EE;
  for (int l = 0; l < 3; ++l) {
    zero_kernel<<<(TT * NN * DH) / 256, 256, 0, stream>>>(agg);
    msg_kernel<<<(TT * (EE / 16)) / 8, 256, 0, stream>>>(
        h16, esrc, edst, m1w + (size_t)l * 2 * DH * DH, msg1_b + l * DH,
        m2w + (size_t)l * DH * DH, msg2_b + l * DH, agg);
    upd_kernel<<<(TT * (NN / 16)) / 8, 256, 0, stream>>>(
        h16, agg, u1w + (size_t)l * 2 * DH * DH, upd1_b + l * DH,
        u2w + (size_t)l * DH * DH, upd2_b + l * DH, h32);
  }

  ro_kernel<<<(TT * NN) / 256, 256, 0, stream>>>(h32, ro_w, ro_b, (float*)d_out);
}
